// DepthFormer_72988674228806
// MI455X (gfx1250) — compile-verified, matrix operand order checked by reading, not run on hardware
//
#include <hip/hip_runtime.h>
#include <hip/hip_bf16.h>
#include <math.h>
#include <stdint.h>

typedef __attribute__((ext_vector_type(16))) _Float16 v16h;
typedef __attribute__((ext_vector_type(8)))  float    v8f;

#define TT     16384      // total tokens (2 * 32 * 256)
#define DIMC   192
#define HEADS  3
#define DHEAD  64
#define INNER  192
#define QKVW   576        // 3 * INNER
#define POSN   8192

// ---------------------------------------------------------------------------
// Fragment helpers (CDNA5 WMMA 16x16x32 f16 layouts, ISA 7.12.2)
// lane<16: K {0..7,16..23}; lane>=16: K {8..15,24..31}; row = lane & 15
// ---------------------------------------------------------------------------
__device__ __forceinline__ v16h frag_from_lds_h(const _Float16* base, int ld, int lane) {
  int r  = lane & 15;
  int ko = (lane & 16) ? 8 : 0;
  v16h f;
#pragma unroll
  for (int j = 0; j < 8; ++j) {
    f[j]     = base[r * ld + ko + j];
    f[8 + j] = base[r * ld + 16 + ko + j];
  }
  return f;
}

__device__ __forceinline__ v16h frag_from_lds_f32(const float* base, int ld, int lane) {
  int r  = lane & 15;
  int ko = (lane & 16) ? 8 : 0;
  v16h f;
#pragma unroll
  for (int j = 0; j < 8; ++j) {
    f[j]     = (_Float16)base[r * ld + ko + j];
    f[8 + j] = (_Float16)base[r * ld + 16 + ko + j];
  }
  return f;
}

// depth attention:  group g = b*256 + n,  token(i) = (b*32 + i)*256 + n
// spatial attention: group g = b*32 + f,  token(i) = g*256 + i
__device__ __forceinline__ int attn_token(int g, int i, int mode) {
  if (mode == 0) {
    int b = g >> 8, nn = g & 255;
    return ((b << 5) + i) * 256 + nn;
  }
  return (g << 8) + i;
}

// ---------------------------------------------------------------------------
// 1) patchify:  volume (2,3,32,128,128) -> P (16384 x 192), col = p1*24+p2*3+ch
// ---------------------------------------------------------------------------
__global__ void patchify(const float* __restrict__ vol, float* __restrict__ P) {
  int idx = blockIdx.x * 256 + threadIdx.x;
  if (idx >= TT * DIMC) return;
  int t = idx / DIMC, col = idx - t * DIMC;
  int ch = col % 3, p2 = (col / 3) & 7, p1 = col / 24;
  int b = t >> 13, rem = t & 8191;
  int d = rem >> 8, sp = rem & 255;
  int hh = sp >> 4, ww = sp & 15;
  int Hc = hh * 8 + p1, Wc = ww * 8 + p2;
  P[idx] = vol[(((size_t)(b * 3 + ch) * 32 + d) * 128 + Hc) * 128 + Wc];
}

// ---------------------------------------------------------------------------
// 2) Generic tiled WMMA GEMM: C[M,N] = A[M,K] @ B[K,N] (+bias) (+residual)
//    block = 256 threads = 8 waves; tile 64x64; wave = 16x32 (2 accumulators)
//    A tile staged via GLOBAL_LOAD_ASYNC_TO_LDS_B128 (ASYNCcnt path)
// ---------------------------------------------------------------------------
template <bool HAS_BIAS, bool HAS_RES>
__global__ __launch_bounds__(256) void wmma_gemm_f32(
    const float* __restrict__ A, int lda,
    const float* __restrict__ B, int ldb,
    float* __restrict__ C, int ldc,
    int M, int N, int K,
    const float* __restrict__ bias,
    const float* __restrict__ residual, int ldr, int rmask) {
  __shared__ float    As[64][36];   // f32 A tile, rows padded to 144B (16B mult.)
  __shared__ _Float16 Bs[64][34];   // transposed (n, k) f16
  int tid = threadIdx.x, wave = tid >> 5, lane = tid & 31;
  int bm = blockIdx.x * 64, bn = blockIdx.y * 64;
  int wm  = (wave >> 1) << 4;       // 0,16,32,48
  int wn0 = (wave & 1) << 5;        // 0,32
  v8f acc0 = {}, acc1 = {};
  uint32_t as_base = (uint32_t)(uintptr_t)&As[0][0];  // flat->LDS offset (addr[31:0])
  for (int k0 = 0; k0 < K; k0 += 32) {
    // A tile 64x32 f32: 512 x 16B chunks, async copied straight into LDS
#pragma unroll
    for (int i = 0; i < 2; ++i) {
      int idx = tid + i * 256;                  // 0..511
      int r = idx >> 3, c = (idx & 7) << 2;     // row, float col (16B chunks)
      const float* gsrc = A + (size_t)(bm + r) * lda + (k0 + c);
      uint32_t dst = as_base + (uint32_t)(r * 36 + c) * 4u;
      asm volatile("global_load_async_to_lds_b128 %0, %1, off"
                   :: "v"(dst), "v"(gsrc) : "memory");
    }
    // B tile 32x64 -> Bs[n][k] (convert + transpose)
#pragma unroll
    for (int i = 0; i < 8; ++i) {
      int idx = tid + i * 256;                  // 0..2047
      int r = idx >> 5, c = idx & 31;           // r = n local, c = k local
      Bs[r][c] = (_Float16)B[(size_t)(k0 + c) * ldb + bn + r];
    }
    asm volatile("s_wait_asynccnt 0" ::: "memory");
    __syncthreads();
    v16h a  = frag_from_lds_f32(&As[wm][0], 36, lane);
    v16h b0 = frag_from_lds_h(&Bs[wn0][0], 34, lane);
    v16h b1 = frag_from_lds_h(&Bs[wn0 + 16][0], 34, lane);
    acc0 = __builtin_amdgcn_wmma_f32_16x16x32_f16(false, a, false, b0,
                                                  (short)0, acc0, false, false);
    acc1 = __builtin_amdgcn_wmma_f32_16x16x32_f16(false, a, false, b1,
                                                  (short)0, acc1, false, false);
    __syncthreads();
  }
  int r0 = (lane & 16) ? 8 : 0, cn = lane & 15;
#pragma unroll
  for (int half = 0; half < 2; ++half) {
    v8f acc = half ? acc1 : acc0;
    int gn = bn + wn0 + half * 16 + cn;
    float bv = HAS_BIAS ? bias[gn] : 0.f;
#pragma unroll
    for (int i = 0; i < 8; ++i) {
      int gm = bm + wm + r0 + i;
      float v = acc[i] + bv;
      if (HAS_RES) v += residual[(size_t)(gm & rmask) * ldr + gn];
      C[(size_t)gm * ldc + gn] = v;
    }
  }
}

// ---------------------------------------------------------------------------
// 3) GEGLU GEMM: h1 = A @ W1 + b1 (N=2*Nh);  G[:,j] = h1[:,j]*gelu(h1[:,Nh+j])
// ---------------------------------------------------------------------------
__global__ __launch_bounds__(256) void wmma_gemm_geglu(
    const float* __restrict__ A, const float* __restrict__ B,
    const float* __restrict__ bias, float* __restrict__ G,
    int M, int K, int Nh) {
  __shared__ _Float16 As[64][34];
  __shared__ _Float16 Bs1[32][34];
  __shared__ _Float16 Bs2[32][34];
  int tid = threadIdx.x, wave = tid >> 5, lane = tid & 31;
  int bm = blockIdx.x * 64, bn = blockIdx.y * 32;
  int wm = (wave >> 1) << 4, wn = (wave & 1) << 4;
  int ldb = 2 * Nh;
  v8f acc1 = {}, acc2 = {};
  for (int k0 = 0; k0 < K; k0 += 32) {
#pragma unroll
    for (int i = 0; i < 8; ++i) {
      int idx = tid + i * 256, r = idx >> 5, c = idx & 31;
      As[r][c] = (_Float16)A[(size_t)(bm + r) * K + k0 + c];
    }
#pragma unroll
    for (int i = 0; i < 4; ++i) {
      int idx = tid + i * 256, r = idx >> 5, c = idx & 31;
      Bs1[r][c] = (_Float16)B[(size_t)(k0 + c) * ldb + bn + r];
      Bs2[r][c] = (_Float16)B[(size_t)(k0 + c) * ldb + Nh + bn + r];
    }
    __syncthreads();
    v16h a  = frag_from_lds_h(&As[wm][0], 34, lane);
    v16h b1 = frag_from_lds_h(&Bs1[wn][0], 34, lane);
    v16h b2 = frag_from_lds_h(&Bs2[wn][0], 34, lane);
    acc1 = __builtin_amdgcn_wmma_f32_16x16x32_f16(false, a, false, b1,
                                                  (short)0, acc1, false, false);
    acc2 = __builtin_amdgcn_wmma_f32_16x16x32_f16(false, a, false, b2,
                                                  (short)0, acc2, false, false);
    __syncthreads();
  }
  int r0 = (lane & 16) ? 8 : 0, cn = lane & 15;
  int gn = bn + wn + cn;
  float ba = bias[gn], bg = bias[Nh + gn];
#pragma unroll
  for (int i = 0; i < 8; ++i) {
    float av = acc1[i] + ba;
    float gv = acc2[i] + bg;
    float gel = 0.5f * gv * (1.f + erff(gv * 0.70710678118654752f));
    G[(size_t)(bm + wm + r0 + i) * Nh + gn] = av * gel;
  }
}

// ---------------------------------------------------------------------------
// 4) ScaleNorm: y = x / max(||x||,eps) * g  (one wave per token, 192 elems)
// ---------------------------------------------------------------------------
__global__ __launch_bounds__(256) void scalenorm_k(
    const float* __restrict__ x, float* __restrict__ y,
    const float* __restrict__ g, int T) {
  int wave = threadIdx.x >> 5, lane = threadIdx.x & 31;
  int t = blockIdx.x * 8 + wave;
  if (t >= T) return;
  const float* row = x + (size_t)t * DIMC;
  float v[6], s = 0.f;
#pragma unroll
  for (int i = 0; i < 6; ++i) { v[i] = row[lane + 32 * i]; s += v[i] * v[i]; }
#pragma unroll
  for (int off = 16; off > 0; off >>= 1) s += __shfl_xor(s, off, 32);
  float scale = g[0] / fmaxf(sqrtf(s), 1e-5f);
  float* out = y + (size_t)t * DIMC;
#pragma unroll
  for (int i = 0; i < 6; ++i) out[lane + 32 * i] = v[i] * scale;
}

// ---------------------------------------------------------------------------
// 5) Attention scores: S = alpha * Q @ K^T  per (group, head); wave per tile
// ---------------------------------------------------------------------------
__global__ __launch_bounds__(128) void attn_scores(
    const float* __restrict__ qkv, float* __restrict__ S,
    int L, int mode, float alpha) {
  int gh = blockIdx.y;
  int g = gh / HEADS, h = gh - g * HEADS;
  int wave = threadIdx.x >> 5, lane = threadIdx.x & 31;
  int tiles_n = L >> 4;
  int tile = blockIdx.x * 4 + wave;
  int tm = (tile / tiles_n) << 4, tn = (tile % tiles_n) << 4;
  int r = lane & 15, ko = (lane & 16) ? 8 : 0;
  const float* qrow = qkv + (size_t)attn_token(g, tm + r, mode) * QKVW + h * DHEAD;
  const float* krow = qkv + (size_t)attn_token(g, tn + r, mode) * QKVW + INNER + h * DHEAD;
  v8f acc = {};
#pragma unroll
  for (int k0 = 0; k0 < DHEAD; k0 += 32) {
    v16h a, b;
#pragma unroll
    for (int j = 0; j < 8; ++j) {
      a[j]     = (_Float16)qrow[k0 + ko + j];
      a[8 + j] = (_Float16)qrow[k0 + 16 + ko + j];
      b[j]     = (_Float16)krow[k0 + ko + j];
      b[8 + j] = (_Float16)krow[k0 + 16 + ko + j];
    }
    acc = __builtin_amdgcn_wmma_f32_16x16x32_f16(false, a, false, b,
                                                 (short)0, acc, false, false);
  }
  int r0 = (lane & 16) ? 8 : 0, cn = lane & 15;
  float* Srow = S + (size_t)gh * L * L;
#pragma unroll
  for (int i = 0; i < 8; ++i)
    Srow[(size_t)(tm + r0 + i) * L + tn + cn] = alpha * acc[i];
}

// ---------------------------------------------------------------------------
// 6) Exact 1.5-entmax over a row (block per row, blockDim = L, L in {32,256})
// ---------------------------------------------------------------------------
__global__ void entmax_rows(float* __restrict__ S, int L) {
  __shared__ float zs[256], c1[256], c2[256], tau_s[256], red[256];
  int i = threadIdx.x;
  float* row = S + (size_t)blockIdx.x * L;
  float z = row[i];
  red[i] = z; __syncthreads();
  for (int s = L >> 1; s > 0; s >>= 1) {
    if (i < s) red[i] = fmaxf(red[i], red[i + s]);
    __syncthreads();
  }
  z -= red[0];
  __syncthreads();
  zs[i] = z; __syncthreads();
  // bitonic sort descending
  for (int k = 2; k <= L; k <<= 1)
    for (int j = k >> 1; j > 0; j >>= 1) {
      int ixj = i ^ j;
      if (ixj > i) {
        float a = zs[i], b = zs[ixj];
        bool desc = ((i & k) == 0);
        if (desc ? (a < b) : (a > b)) { zs[i] = b; zs[ixj] = a; }
      }
      __syncthreads();
    }
  // inclusive scans of zs and zs^2
  c1[i] = zs[i]; c2[i] = zs[i] * zs[i]; __syncthreads();
  for (int off = 1; off < L; off <<= 1) {
    float a1 = (i >= off) ? c1[i - off] : 0.f;
    float a2 = (i >= off) ? c2[i - off] : 0.f;
    __syncthreads();
    c1[i] += a1; c2[i] += a2;
    __syncthreads();
  }
  float kk = (float)(i + 1);
  float mean = c1[i] / kk, meansq = c2[i] / kk;
  float ss = kk * (meansq - mean * mean);
  float delta = (1.f - ss) / kk;
  float tau = mean - sqrtf(fmaxf(delta, 0.f));
  tau_s[i] = tau;
  red[i] = (tau <= zs[i]) ? 1.f : 0.f; __syncthreads();
  for (int s = L >> 1; s > 0; s >>= 1) {
    if (i < s) red[i] += red[i + s];
    __syncthreads();
  }
  int support = (int)red[0];
  float p = fmaxf(z - tau_s[support - 1], 0.f);
  row[i] = p * p;
}

// ---------------------------------------------------------------------------
// 7) O = P @ V per (group, head); writes head slice of tmp; wave per tile
// ---------------------------------------------------------------------------
__global__ __launch_bounds__(128) void attn_pv(
    const float* __restrict__ qkv, const float* __restrict__ P,
    float* __restrict__ out, int L, int mode) {
  int gh = blockIdx.y;
  int g = gh / HEADS, h = gh - g * HEADS;
  int wave = threadIdx.x >> 5, lane = threadIdx.x & 31;
  int tile = blockIdx.x * 4 + wave;
  int tm = (tile >> 2) << 4, tn = (tile & 3) << 4;
  int r = lane & 15, ko = (lane & 16) ? 8 : 0;
  const float* Pm = P + (size_t)gh * L * L;
  v8f acc = {};
  for (int k0 = 0; k0 < L; k0 += 32) {
    v16h a, b;
    const float* prow = Pm + (size_t)(tm + r) * L + k0;
#pragma unroll
    for (int j = 0; j < 8; ++j) {
      a[j]     = (_Float16)prow[ko + j];
      a[8 + j] = (_Float16)prow[16 + ko + j];
      int k1 = k0 + ko + j, k2 = k0 + 16 + ko + j;
      b[j]     = (_Float16)qkv[(size_t)attn_token(g, k1, mode) * QKVW + 2 * INNER + h * DHEAD + tn + r];
      b[8 + j] = (_Float16)qkv[(size_t)attn_token(g, k2, mode) * QKVW + 2 * INNER + h * DHEAD + tn + r];
    }
    acc = __builtin_amdgcn_wmma_f32_16x16x32_f16(false, a, false, b,
                                                 (short)0, acc, false, false);
  }
  int r0 = (lane & 16) ? 8 : 0, cn = lane & 15;
#pragma unroll
  for (int i = 0; i < 8; ++i) {
    int ti = attn_token(g, tm + r0 + i, mode);
    out[(size_t)ti * DIMC + h * DHEAD + tn + cn] = acc[i];
  }
}

// ---------------------------------------------------------------------------
// Host orchestration
// ---------------------------------------------------------------------------
extern "C" void kernel_launch(void* const* d_in, const int* in_sizes, int n_in,
                              void* d_out, int out_size, void* d_ws, size_t ws_size,
                              hipStream_t stream) {
  (void)in_sizes; (void)n_in; (void)out_size; (void)ws_size;
  const float* vol     = (const float*)d_in[0];
  const float* w_patch = (const float*)d_in[1];
  const float* b_patch = (const float*)d_in[2];
  const float* pos_emb = (const float*)d_in[3];

  float* x    = (float*)d_ws;                 // 16384*192
  float* y    = x    + (size_t)TT * DIMC;     // 16384*192
  float* tmp  = y    + (size_t)TT * DIMC;     // 16384*192
  float* qkv  = tmp  + (size_t)TT * DIMC;     // 16384*576
  float* attn = qkv  + (size_t)TT * QKVW;     // max(192*256*256, 16384*768)

  const float alpha = 0.5f * 0.125f;          // 0.5 * DIM_HEAD^-0.5

  // patch embed: P -> y, x = y @ w_patch + b_patch + pos_emb[t & 8191]
  patchify<<<(TT * DIMC + 255) / 256, 256, 0, stream>>>(vol, y);
  wmma_gemm_f32<true, true><<<dim3(TT / 64, DIMC / 64), 256, 0, stream>>>(
      y, DIMC, w_patch, DIMC, x, DIMC, TT, DIMC, DIMC, b_patch,
      pos_emb, DIMC, POSN - 1);

  for (int l = 0; l < 4; ++l) {
    int base = 4 + l * 13;
    const float* g_dep    = (const float*)d_in[base + 0];
    const float* dep_wqkv = (const float*)d_in[base + 1];
    const float* dep_wout = (const float*)d_in[base + 2];
    const float* dep_bout = (const float*)d_in[base + 3];
    const float* g_spa    = (const float*)d_in[base + 4];
    const float* spa_wqkv = (const float*)d_in[base + 5];
    const float* spa_wout = (const float*)d_in[base + 6];
    const float* spa_bout = (const float*)d_in[base + 7];
    const float* g_ff     = (const float*)d_in[base + 8];
    const float* w_ff1    = (const float*)d_in[base + 9];
    const float* b_ff1    = (const float*)d_in[base + 10];
    const float* w_ff2    = (const float*)d_in[base + 11];
    const float* b_ff2    = (const float*)d_in[base + 12];

    // ---- depth attention (G = 512 groups, L = 32) ----
    scalenorm_k<<<TT / 8, 256, 0, stream>>>(x, y, g_dep, TT);
    wmma_gemm_f32<false, false><<<dim3(TT / 64, QKVW / 64), 256, 0, stream>>>(
        y, DIMC, dep_wqkv, QKVW, qkv, QKVW, TT, QKVW, DIMC,
        nullptr, nullptr, 0, 0);
    attn_scores<<<dim3(1, 512 * HEADS), 128, 0, stream>>>(qkv, attn, 32, 0, alpha);
    entmax_rows<<<512 * HEADS * 32, 32, 0, stream>>>(attn, 32);
    attn_pv<<<dim3(2, 512 * HEADS), 128, 0, stream>>>(qkv, attn, tmp, 32, 0);
    wmma_gemm_f32<true, true><<<dim3(TT / 64, DIMC / 64), 256, 0, stream>>>(
        tmp, DIMC, dep_wout, DIMC, x, DIMC, TT, DIMC, DIMC,
        dep_bout, x, DIMC, TT - 1);

    // ---- spatial attention (G = 64 groups, L = 256) ----
    scalenorm_k<<<TT / 8, 256, 0, stream>>>(x, y, g_spa, TT);
    wmma_gemm_f32<false, false><<<dim3(TT / 64, QKVW / 64), 256, 0, stream>>>(
        y, DIMC, spa_wqkv, QKVW, qkv, QKVW, TT, QKVW, DIMC,
        nullptr, nullptr, 0, 0);
    attn_scores<<<dim3(64, 64 * HEADS), 128, 0, stream>>>(qkv, attn, 256, 1, alpha);
    entmax_rows<<<64 * HEADS * 256, 256, 0, stream>>>(attn, 256);
    attn_pv<<<dim3(16, 64 * HEADS), 128, 0, stream>>>(qkv, attn, tmp, 256, 1);
    wmma_gemm_f32<true, true><<<dim3(TT / 64, DIMC / 64), 256, 0, stream>>>(
        tmp, DIMC, spa_wout, DIMC, x, DIMC, TT, DIMC, DIMC,
        spa_bout, x, DIMC, TT - 1);

    // ---- GEGLU feed-forward ----
    scalenorm_k<<<TT / 8, 256, 0, stream>>>(x, y, g_ff, TT);
    wmma_gemm_geglu<<<dim3(TT / 64, 768 / 32), 256, 0, stream>>>(
        y, w_ff1, b_ff1, attn /* G: 16384x768 */, TT, DIMC, 768);
    wmma_gemm_f32<true, true><<<dim3(TT / 64, DIMC / 64), 256, 0, stream>>>(
        attn, 768, w_ff2, DIMC, x, DIMC, TT, DIMC, 768,
        b_ff2, x, DIMC, TT - 1);
  }

  hipMemcpyAsync(d_out, x, (size_t)TT * DIMC * sizeof(float),
                 hipMemcpyDeviceToDevice, stream);
}
